// StackedBRNN_15247133901006
// MI455X (gfx1250) — matrix operand order repaired and checked by reading, not compiled, not run on hardware
//
#include <hip/hip_runtime.h>
#include <hip/hip_bf16.h>

// ---------------------------------------------------------------------------
// Stacked bidirectional SRU on MI455X (gfx1250):
//   3 x [ bf16-WMMA GEMM (TDM-staged tiles) -> fused time-scan recurrence ]
//   x:(B=32,T=1024,D=512) fp32, W:(512,2,3,256) fp32, b:(2,2,256) fp32
//   u = h @ W : M = T*B = 32768, N = 1536, K = 512
// Operands pre-converted to bf16 once (W transposed to N-major) so the GEMM
// K-loop is pure data movement + v_wmma. Tiles staged LDS-side by the Tensor
// Data Mover with hardware padding (64B row + 16B pad = 80B stride).
// ---------------------------------------------------------------------------

typedef __attribute__((ext_vector_type(16))) __bf16 v16bf;
typedef __attribute__((ext_vector_type(8)))  float  v8f;
typedef __attribute__((ext_vector_type(4)))  unsigned int v4u;
typedef __attribute__((ext_vector_type(8)))  int v8i;
typedef __attribute__((ext_vector_type(4)))  int v4i;

#if defined(__has_builtin)
#if __has_builtin(__builtin_amdgcn_tensor_load_to_lds) && \
    __has_builtin(__builtin_amdgcn_s_wait_tensorcnt)
#define SRU_USE_TDM 1
#endif
#endif
#ifndef SRU_USE_TDM
#define SRU_USE_TDM 0
#endif

union FragU {
  v16bf v;
  uint4 q[2];
};

__device__ __forceinline__ unsigned short f32_to_bf16(float f) {
  union { float f; unsigned int u; } x;
  x.f = f;
  unsigned int r = x.u + 0x7FFFu + ((x.u >> 16) & 1u);  // round-to-nearest-even
  return (unsigned short)(r >> 16);
}
__device__ __forceinline__ unsigned int pack2(unsigned short lo, unsigned short hi) {
  return (unsigned int)lo | ((unsigned int)hi << 16);
}

#define BM 128
#define BN 128
#define BK 32
#define LDSS 40   // LDS row stride in halves: 64B data + 16B pad = 80B (16B aligned,
                  // 20-bank stride -> conflict-free 16-lane b128 fragment reads)

#if SRU_USE_TDM
// Issue one TDM 2D tile load: tile_dim0=32 halves (64B), tile_dim1=128 rows,
// from a bf16 tensor with row stride `rowElems`, into LDS at `lds_off` with
// hardware padding: 4 DWORDs pad after every 16 DWORDs -> 80B LDS row stride.
__device__ __forceinline__ void tdm_load_tile(const unsigned short* gsrc,
                                              unsigned lds_off,
                                              unsigned rowElems,
                                              unsigned totalRows) {
  unsigned long long ga = (unsigned long long)(size_t)gsrc;
  v4u g0;
  g0.x = 1u;                                            // count=1 (valid), no gather
  g0.y = lds_off;                                       // LDS byte address
  g0.z = (unsigned)(ga & 0xFFFFFFFFu);                  // global_addr[31:0]
  g0.w = (unsigned)((ga >> 32) & 0x1FFFFFFu) | (2u << 30);  // addr[56:32], type=2

  const unsigned long long td0 = rowElems;              // tensor_dim0 (elements)
  const unsigned long long td1 = totalRows;             // tensor_dim1
  const unsigned long long s0  = rowElems;              // tensor_dim0_stride
  v8i g1;
  g1[0] = (int)((1u << 16)      /* data_size = 2 bytes   */
              | (1u << 20)      /* pad_enable            */
              | (3u << 22)      /* pad_interval: 16 DWORDs */
              | (3u << 25));    /* pad_amount:   4 DWORDs  */
  g1[1] = (int)((td0 & 0xFFFFu) << 16);                             // abar=0 | td0 lo
  g1[2] = (int)(((td0 >> 16) & 0xFFFFu) | ((td1 & 0xFFFFu) << 16)); // td0 hi | td1 lo
  g1[3] = (int)(((td1 >> 16) & 0xFFFFu) | (32u << 16));             // td1 hi | tile_dim0
  g1[4] = (int)(128u);                                              // tile_dim1 | tile_dim2=0
  g1[5] = (int)(s0 & 0xFFFFFFFFu);
  g1[6] = (int)((s0 >> 32) & 0xFFFFu);                              // | dim1_stride=0
  g1[7] = 0;
  v4i gz = {0, 0, 0, 0};
#if defined(__clang_major__) && __clang_major__ >= 23
  v8i gz8 = {0, 0, 0, 0, 0, 0, 0, 0};
  __builtin_amdgcn_tensor_load_to_lds(g0, g1, gz, gz, gz8, 0);
#else
  __builtin_amdgcn_tensor_load_to_lds(g0, g1, gz, gz, 0);
#endif
}
#endif  // SRU_USE_TDM

// ---------------------------------------------------------------------------
// GEMM: U[M=32768 x N=1536] = Abf[M x 512] * WbfT[N x 512]^T   (both bf16)
// ---------------------------------------------------------------------------
__global__ __launch_bounds__(256) void sru_gemm_wmma(
    const unsigned short* __restrict__ Abf,    // bf16, row-major M x 512
    const unsigned short* __restrict__ WbfT,   // bf16, row-major N x 512 (W transposed)
    float* __restrict__ U) {
  __shared__ unsigned short ldsA[BM * LDSS];   // [m][k]
  __shared__ unsigned short ldsB[BN * LDSS];   // [n][k]

  const int tid    = threadIdx.x;
  const int lane   = tid & 31;
  const int wave   = tid >> 5;
  const int waveM  = wave >> 1;          // 0..3 -> 32-row strip
  const int waveN  = wave & 1;           // 0..1 -> 64-col strip
  const int blockN = blockIdx.x;         // 0..11
  const int blockM = blockIdx.y;         // 0..255

  v8f acc[2][4];
#pragma unroll
  for (int i = 0; i < 2; ++i)
#pragma unroll
    for (int j = 0; j < 4; ++j) acc[i][j] = {};

  const unsigned ldsA_off = (unsigned)(size_t)(&ldsA[0]);
  const unsigned ldsB_off = (unsigned)(size_t)(&ldsB[0]);
  (void)ldsA_off; (void)ldsB_off;

  for (int k0 = 0; k0 < 512; k0 += BK) {
    __syncthreads();  // previous iteration's fragment reads done
#if SRU_USE_TDM
    if (wave == 0) {  // TDM ignores EXEC; gate by scalar branch, one wave issues
      tdm_load_tile(Abf  + (size_t)blockM * BM * 512 + k0, ldsA_off, 512u, 32768u);
      tdm_load_tile(WbfT + (size_t)blockN * BN * 512 + k0, ldsB_off, 512u, 1536u);
      __builtin_amdgcn_s_wait_tensorcnt(0);
    }
#else
    // Fallback: manual staging, 16B chunks (tile = 128 rows x 64B = 512 chunks)
#pragma unroll
    for (int i = 0; i < 2; ++i) {
      const int cid = tid + i * 256;
      const int row = cid >> 2;
      const int cq  = (cid & 3) * 8;   // halves
      *reinterpret_cast<uint4*>(&ldsA[row * LDSS + cq]) =
          *reinterpret_cast<const uint4*>(Abf + (size_t)(blockM * BM + row) * 512 + k0 + cq);
      *reinterpret_cast<uint4*>(&ldsB[row * LDSS + cq]) =
          *reinterpret_cast<const uint4*>(WbfT + (size_t)(blockN * BN + row) * 512 + k0 + cq);
    }
#endif
    __syncthreads();  // staged tile visible to all waves

    // Fragments per ISA 16-bit A/B layouts (two ds_load_b128 each)
    FragU afr[2], bfr[4];
    const int klo  = (lane >> 4) * 8;   // A: lanes 0-15 K0..7/16..23, 16-31 K8..15/24..31
    const int mrow = waveM * 32 + (lane & 15);
#pragma unroll
    for (int mi = 0; mi < 2; ++mi) {
      const int m = mrow + mi * 16;
      afr[mi].q[0] = *reinterpret_cast<const uint4*>(&ldsA[m * LDSS + klo]);
      afr[mi].q[1] = *reinterpret_cast<const uint4*>(&ldsA[m * LDSS + klo + 16]);
    }
    const int kbf = (lane >> 4) * 8;    // B: lanes 0-15 K0..7/16..23, 16-31 K8..15/24..31
#pragma unroll
    for (int ni = 0; ni < 4; ++ni) {
      const int n = waveN * 64 + ni * 16 + (lane & 15);
      bfr[ni].q[0] = *reinterpret_cast<const uint4*>(&ldsB[n * LDSS + kbf]);
      bfr[ni].q[1] = *reinterpret_cast<const uint4*>(&ldsB[n * LDSS + kbf + 16]);
    }
#pragma unroll
    for (int mi = 0; mi < 2; ++mi)
#pragma unroll
      for (int ni = 0; ni < 4; ++ni)
        acc[mi][ni] = __builtin_amdgcn_wmma_f32_16x16x32_bf16(
            false, afr[mi].v, false, bfr[ni].v, (short)0, acc[mi][ni],
            false, false);
  }

  // C layout: lanes 0-15 -> M=i, lanes 16-31 -> M=8+i; N = lane&15
  const int rBase = blockM * BM + waveM * 32 + ((lane >> 4) * 8);
  const int cBase = blockN * BN + waveN * 64 + (lane & 15);
#pragma unroll
  for (int mi = 0; mi < 2; ++mi)
#pragma unroll
    for (int ni = 0; ni < 4; ++ni) {
      const size_t cc = (size_t)(cBase + ni * 16);
#pragma unroll
      for (int i = 0; i < 8; ++i)
        U[(size_t)(rBase + mi * 16 + i) * 1536 + cc] = acc[mi][ni][i];
    }
}

// ---------------------------------------------------------------------------
// SRU recurrence, fused with next layer's bf16 operand production.
// One block per (batch b, direction k); one thread per channel; dir 1 walks
// t backwards (== the reference's flip/flip-back).
// ---------------------------------------------------------------------------
__global__ __launch_bounds__(256) void sru_scan(
    const float* __restrict__ U,
    const float* __restrict__ Xr, int xr_sT, int xr_sB,
    const float* __restrict__ bias,
    float* __restrict__ Out, int out_sT, int out_sB,
    unsigned short* __restrict__ OutBf) {   // bf16 (T,B,D) for next GEMM, or null
  const int b  = blockIdx.x >> 1;
  const int k  = blockIdx.x & 1;
  const int hh = threadIdx.x;

  const float bf = bias[k * 512 + hh];        // b[k][0][hh]
  const float br = bias[k * 512 + 256 + hh];  // b[k][1][hh]
  const int resCol = k * 256 + hh;

  float c = 0.0f;
  for (int s = 0; s < 1024; ++s) {
    const int t = (k == 0) ? s : (1023 - s);
    const float* up = U + (size_t)(t * 32 + b) * 1536 + k * 768 + hh;
    const float xt = up[0];
    const float fu = up[256];
    const float ru = up[512];
    const float f = 1.0f / (1.0f + __expf(-(fu + bf)));
    const float r = 1.0f / (1.0f + __expf(-(ru + br)));
    c = f * c + (1.0f - f) * xt;
    const float xr = Xr[(size_t)t * xr_sT + (size_t)b * xr_sB + resCol];
    const float h = r * tanhf(c) + (1.0f - r) * xr;
    Out[(size_t)t * out_sT + (size_t)b * out_sB + resCol] = h;
    if (OutBf) OutBf[(size_t)(t * 32 + b) * 512 + resCol] = f32_to_bf16(h);
  }
}

// ---------------------------------------------------------------------------
// One-time operand conversions.
// ---------------------------------------------------------------------------
// x (B,T,D) fp32 -> bf16 (T,B,D) flat M x 512
__global__ __launch_bounds__(256) void cvt_act_bf16(
    const float* __restrict__ src, int sT, int sB, unsigned short* __restrict__ dst) {
  const int idx = blockIdx.x * 256 + threadIdx.x;   // one float4 per thread
  const int e4  = idx * 4;
  const int r   = e4 >> 9;           // GEMM row = t*32 + b
  const int col = e4 & 511;
  const int t   = r >> 5;
  const int b   = r & 31;
  const float4 v = *reinterpret_cast<const float4*>(src + (size_t)t * sT + (size_t)b * sB + col);
  uint2 p;
  p.x = pack2(f32_to_bf16(v.x), f32_to_bf16(v.y));
  p.y = pack2(f32_to_bf16(v.z), f32_to_bf16(v.w));
  *reinterpret_cast<uint2*>(dst + (size_t)r * 512 + col) = p;
}

// W (512 x 1536) fp32 -> bf16 transposed (1536 x 512)
__global__ __launch_bounds__(256) void cvt_w_transpose(
    const float* __restrict__ W, unsigned short* __restrict__ WT) {
  const int n = blockIdx.x * 256 + threadIdx.x;   // 0..1535 (coalesced read)
  const int d = blockIdx.y;                       // 0..511
  WT[(size_t)n * 512 + d] = f32_to_bf16(W[(size_t)d * 1536 + n]);
}

// ---------------------------------------------------------------------------
extern "C" void kernel_launch(void* const* d_in, const int* in_sizes, int n_in,
                              void* d_out, int out_size, void* d_ws, size_t ws_size,
                              hipStream_t stream) {
  (void)in_sizes; (void)n_in; (void)out_size; (void)ws_size;

  const float* x  = (const float*)d_in[0];
  const float* Wl[3] = {(const float*)d_in[1], (const float*)d_in[3], (const float*)d_in[5]};
  const float* bl[3] = {(const float*)d_in[2], (const float*)d_in[4], (const float*)d_in[6]};
  float* out = (float*)d_out;

  const size_t M = 32768;  // T*B
  char* ws = (char*)d_ws;
  float* u  = (float*)ws;                                   // 192 MiB
  float* h0 = (float*)(ws + M * 1536 * 4);                  //  64 MiB
  float* h1 = h0 + M * 512;                                 //  64 MiB
  unsigned short* ab0 = (unsigned short*)(h1 + M * 512);    //  32 MiB
  unsigned short* ab1 = ab0 + M * 512;                      //  32 MiB
  unsigned short* wbf = ab1 + M * 512;                      // 1.5 MiB

  const dim3 gGrid(12, 256), blk(256);
  const dim3 sGrid(64);
  const dim3 caGrid((unsigned)(M * 512 / (4 * 256)));       // 16384
  const dim3 cwGrid(6, 512);

  const int xT = 512, xB = 1024 * 512;    // (B,T,D) strides
  const int hT = 32 * 512, hB = 512;      // (T,B,D) strides

  cvt_act_bf16<<<caGrid, blk, 0, stream>>>(x, xT, xB, ab0);

  // layer 0: x -> h0 (+ab1 bf16)
  cvt_w_transpose<<<cwGrid, blk, 0, stream>>>(Wl[0], wbf);
  sru_gemm_wmma<<<gGrid, blk, 0, stream>>>(ab0, wbf, u);
  sru_scan<<<sGrid, blk, 0, stream>>>(u, x, xT, xB, bl[0], h0, hT, hB, ab1);

  // layer 1: h0 -> h1 (+ab0 bf16)
  cvt_w_transpose<<<cwGrid, blk, 0, stream>>>(Wl[1], wbf);
  sru_gemm_wmma<<<gGrid, blk, 0, stream>>>(ab1, wbf, u);
  sru_scan<<<sGrid, blk, 0, stream>>>(u, h0, hT, hB, bl[1], h1, hT, hB, ab0);

  // layer 2: h1 -> d_out (B,T,D)
  cvt_w_transpose<<<cwGrid, blk, 0, stream>>>(Wl[2], wbf);
  sru_gemm_wmma<<<gGrid, blk, 0, stream>>>(ab0, wbf, u);
  sru_scan<<<sGrid, blk, 0, stream>>>(u, h1, hT, hB, bl[2], out, xT, xB, nullptr);
}
